// EncoderTransformerLayer_17214228922882
// MI455X (gfx1250) — matrix operand order repaired and checked by reading, not compiled, not run on hardware
//
#include <hip/hip_runtime.h>
#include <math.h>

// ---------------------------------------------------------------------------
// EncoderTransformerLayer for MI455X (gfx1250), bf16 WMMA + TDM pipeline.
// B=4, T=2048, H=1024.  GEMMs: v_wmma_f32_16x16x32_bf16, 64x64 wave tiles,
// 256x128 block tiles, TDM (tensor_load_to_lds) ping-pong LDS staging.
// Workspace: xb@0 (16MB, reused as h1), Wt[5]@16MB (10MB), qb@26MB (16MB,
// reused as attn_out), kb@42MB, vt@58MB -> 74MB total.  d_out doubles as
// per-batch scores(fp32,16MB)+attn(bf16,8MB) scratch (dead by final GEMM).
// ---------------------------------------------------------------------------

typedef __attribute__((ext_vector_type(16))) __bf16 v16bf;
typedef __attribute__((ext_vector_type(8)))  float  v8f;
typedef __attribute__((ext_vector_type(4)))  unsigned int u32x4;
typedef __attribute__((ext_vector_type(8)))  int i32x8;
typedef __attribute__((ext_vector_type(4)))  int i32x4;

#define LDSW 40   // padded LDS row stride in halfwords (64B data + 16B pad)

// Epilogue modes
#define EPI_BF16        0   // outb = acc + bias
#define EPI_BF16_GELU   1   // outb = gelu(acc + bias)
#define EPI_BF16_VT     2   // outb[vT] = acc + bias   (v transposed store)
#define EPI_F32_SCALE   3   // outf = acc * scale      (attention scores)
#define EPI_F32_RESID   4   // outf = acc + bias + residual (final output)
#define EPI_BF16_NOBIAS 5   // outb = acc              (attn @ v)

__device__ __forceinline__ unsigned short f2bf(float f) {
    unsigned int u = __float_as_uint(f);
    u += 0x7FFFu + ((u >> 16) & 1u);   // round-to-nearest-even
    return (unsigned short)(u >> 16);
}

__device__ __forceinline__ float gelu_tanh(float x) {
    float x3 = x * x * x;
    return 0.5f * x * (1.0f + tanhf(0.7978845608028654f * (x + 0.044715f * x3)));
}

// TDM: load a (tile_rows x 32) bf16 tile from row-major src (row stride = K
// elements) starting at (row0, k0) into LDS at byte offset lds_byte, with
// 16B padding after every 64B row => 80B LDS row stride (matches LDSW).
__device__ __forceinline__ void tdm_load_tile(const unsigned short* src,
                                              unsigned int lds_byte,
                                              int rows_total, int K,
                                              int row0, int k0, int tile_rows)
{
    unsigned long long ga =
        (unsigned long long)(size_t)(src + (size_t)row0 * K + k0);
    u32x4 g0;
    g0[0] = 1u;                                        // count=1, no gather
    g0[1] = lds_byte;                                  // lds_addr
    g0[2] = (unsigned int)(ga & 0xFFFFFFFFull);        // global_addr[31:0]
    g0[3] = (unsigned int)((ga >> 32) & 0x1FFFFFFull)  // global_addr[56:32]
          | (2u << 30);                                // type=2 (image)
    i32x8 g1;
    // workgroup_mask=0 | data_size=1(2B)<<16 | pad_en<<20 |
    // pad_interval=3 (16 DWORDs = 64B) <<22 | pad_amount=3 (4 DWORDs=16B) <<25
    g1[0] = (int)((1u << 16) | (1u << 20) | (3u << 22) | (3u << 25));
    g1[1] = (int)(((unsigned)K & 0xFFFFu) << 16);                       // dim0 lo
    g1[2] = (int)((((unsigned)K >> 16) & 0xFFFFu)
          | (((unsigned)rows_total & 0xFFFFu) << 16));                  // dim0 hi | dim1 lo
    g1[3] = (int)(((((unsigned)rows_total) >> 16) & 0xFFFFu)
          | (32u << 16));                                               // dim1 hi | tile_dim0=32
    g1[4] = (int)(unsigned)tile_rows;                                   // tile_dim1 (tile_dim2=0)
    g1[5] = (int)(unsigned)K;                                           // dim0_stride[31:0]
    g1[6] = 0;                                                          // stride hi
    g1[7] = 0;
    i32x4 z4; z4[0] = z4[1] = z4[2] = z4[3] = 0;
    i32x8 z8;
    #pragma unroll
    for (int e = 0; e < 8; ++e) z8[e] = 0;
    __builtin_amdgcn_tensor_load_to_lds(g0, g1, z4, z4, z8, 0);
}

// C[m][n] = sum_k A[m][k]*Bt[n][k].  M%256==0, N%128==0, K%32==0.
template <int EPI>
__global__ __launch_bounds__(256)
void gemm_bf16_wmma(const unsigned short* __restrict__ A,
                    const unsigned short* __restrict__ Bt,
                    int M, int N, int K, float scale,
                    const float* __restrict__ bias,
                    const float* __restrict__ residual,
                    unsigned short* __restrict__ outb,
                    float* __restrict__ outf)
{
    // ping-pong buffers: each = A tile 256xLDSW + B tile 128xLDSW halfwords
    __shared__ alignas(16) unsigned short smem[2][(256 + 128) * LDSW];

    const int tid   = threadIdx.x;
    const int lane  = tid & 31;
    const int wid   = tid >> 5;
    const int waveM = wid >> 1;       // 0..3 -> 64-row strip
    const int waveN = wid & 1;        // 0..1 -> 64-col strip
    const int l15   = lane & 15;
    const int half  = lane >> 4;
    const int m0    = blockIdx.y * 256;
    const int n0    = blockIdx.x * 128;

    // Generic shared-space addr[31:0] == LDS byte offset (ISA 10.2 aperture map)
    const unsigned int lds0 = (unsigned int)(unsigned long long)(void*)&smem[0][0];
    const unsigned int seg  = (256 + 128) * LDSW * 2;   // bytes per buffer
    const unsigned int bOff = 256 * LDSW * 2;           // B tile offset in buffer

    v8f acc[4][4];
    #pragma unroll
    for (int mi = 0; mi < 4; ++mi)
        #pragma unroll
        for (int ni = 0; ni < 4; ++ni)
            #pragma unroll
            for (int e = 0; e < 8; ++e) acc[mi][ni][e] = 0.0f;

    const int nK = K >> 5;

    if (wid == 0) {
        tdm_load_tile(A,  lds0,        M, K, m0, 0, 256);
        tdm_load_tile(Bt, lds0 + bOff, N, K, n0, 0, 128);
        __builtin_amdgcn_s_wait_tensorcnt((short)0);
    }
    __syncthreads();

    int p = 0;
    for (int kt = 0; kt < nK; ++kt) {
        if (wid == 0 && kt + 1 < nK) {   // prefetch next K-slab into other buffer
            unsigned int dst = lds0 + (unsigned)(p ^ 1) * seg;
            tdm_load_tile(A,  dst,        M, K, m0, (kt + 1) << 5, 256);
            tdm_load_tile(Bt, dst + bOff, N, K, n0, (kt + 1) << 5, 128);
        }
        const unsigned short* As = &smem[p][0];
        const unsigned short* Bs = &smem[p][256 * LDSW];

        union Frag { uint4 u[2]; v16bf v; };
        Frag af[4], bf[4];
        // A frag 16x32: lane holds row (lane&15); chunks at half*8, half*8+16
        #pragma unroll
        for (int mi = 0; mi < 4; ++mi) {
            const unsigned short* ap = &As[(waveM * 64 + mi * 16 + l15) * LDSW + half * 8];
            af[mi].u[0] = *(const uint4*)(ap);
            af[mi].u[1] = *(const uint4*)(ap + 16);
        }
        // B frag 32x16: lane holds col (lane&15); chunks at half*8, half*8+16
        #pragma unroll
        for (int ni = 0; ni < 4; ++ni) {
            const unsigned short* bp = &Bs[(waveN * 64 + ni * 16 + l15) * LDSW + half * 8];
            bf[ni].u[0] = *(const uint4*)(bp);
            bf[ni].u[1] = *(const uint4*)(bp + 16);
        }
        #pragma unroll
        for (int mi = 0; mi < 4; ++mi)
            #pragma unroll
            for (int ni = 0; ni < 4; ++ni)
                acc[mi][ni] = __builtin_amdgcn_wmma_f32_16x16x32_bf16(
                    false, af[mi].v, false, bf[ni].v,
                    (short)0, acc[mi][ni], false, false);

        if (wid == 0) __builtin_amdgcn_s_wait_tensorcnt((short)0);
        __syncthreads();
        p ^= 1;
    }

    // Epilogue. C/D layout: VGPR i -> M = i + 8*(lane>>4), N = lane&15.
    int ncol[4];
    float badd[4];
    #pragma unroll
    for (int ni = 0; ni < 4; ++ni) {
        ncol[ni] = n0 + waveN * 64 + ni * 16 + l15;
        badd[ni] = (EPI == EPI_F32_SCALE || EPI == EPI_BF16_NOBIAS)
                 ? 0.0f : bias[ncol[ni]];
    }
    #pragma unroll
    for (int mi = 0; mi < 4; ++mi) {
        const int mbase = m0 + waveM * 64 + mi * 16 + (half ? 8 : 0);
        #pragma unroll
        for (int ni = 0; ni < 4; ++ni) {
            const int n = ncol[ni];
            #pragma unroll
            for (int i = 0; i < 8; ++i) {
                const int m = mbase + i;
                if (EPI == EPI_F32_SCALE) {
                    outf[(size_t)m * N + n] = acc[mi][ni][i] * scale;
                } else if (EPI == EPI_F32_RESID) {
                    size_t idx = (size_t)m * N + n;
                    outf[idx] = acc[mi][ni][i] + badd[ni] + residual[idx];
                } else if (EPI == EPI_BF16_GELU) {
                    outb[(size_t)m * N + n] = f2bf(gelu_tanh(acc[mi][ni][i] + badd[ni]));
                } else if (EPI == EPI_BF16_VT) {
                    // vT[b][h][t]: b=m>>11, t=m&2047, h=n  (T=2048, H=1024)
                    size_t idx = ((size_t)(m >> 11) * 1024 + n) * 2048 + (m & 2047);
                    outb[idx] = f2bf(acc[mi][ni][i] + badd[ni]);
                } else if (EPI == EPI_BF16_NOBIAS) {
                    outb[(size_t)m * N + n] = f2bf(acc[mi][ni][i]);
                } else { // EPI_BF16
                    outb[(size_t)m * N + n] = f2bf(acc[mi][ni][i] + badd[ni]);
                }
            }
        }
    }
}

// Row softmax over 2048 fp32 logits -> bf16 probabilities. One block per row.
__global__ __launch_bounds__(256)
void softmax2048(const float* __restrict__ scores, unsigned short* __restrict__ attn)
{
    __shared__ float red[256];
    const int row = blockIdx.x;
    const int tid = threadIdx.x;
    const float* src = scores + (size_t)row * 2048;

    float v[8];
    float mx = -3.4e38f;
    #pragma unroll
    for (int j = 0; j < 8; ++j) { v[j] = src[j * 256 + tid]; mx = fmaxf(mx, v[j]); }
    red[tid] = mx; __syncthreads();
    for (int s = 128; s > 0; s >>= 1) {
        if (tid < s) red[tid] = fmaxf(red[tid], red[tid + s]);
        __syncthreads();
    }
    mx = red[0]; __syncthreads();

    float sum = 0.0f;
    #pragma unroll
    for (int j = 0; j < 8; ++j) { v[j] = __expf(v[j] - mx); sum += v[j]; }
    red[tid] = sum; __syncthreads();
    for (int s = 128; s > 0; s >>= 1) {
        if (tid < s) red[tid] += red[tid + s];
        __syncthreads();
    }
    const float inv = 1.0f / red[0];

    unsigned short* dst = attn + (size_t)row * 2048;
    #pragma unroll
    for (int j = 0; j < 8; ++j) dst[j * 256 + tid] = f2bf(v[j] * inv);
}

__global__ __launch_bounds__(256)
void cvt_f32_bf16(const float* __restrict__ src, unsigned short* __restrict__ dst, int n)
{
    int i = blockIdx.x * 256 + threadIdx.x;
    if (i < n) dst[i] = f2bf(src[i]);
}

// W[1024][1024] fp32 (row-major [K][N]) -> Wt[1024][1024] bf16 ([N][K])
__global__ __launch_bounds__(256)
void transpose_cvt_1024(const float* __restrict__ W, unsigned short* __restrict__ Wt)
{
    int idx = blockIdx.x * 256 + threadIdx.x;   // 0 .. 1048575
    int k = idx >> 10, n = idx & 1023;
    Wt[(size_t)n * 1024 + k] = f2bf(W[idx]);
}

extern "C" void kernel_launch(void* const* d_in, const int* in_sizes, int n_in,
                              void* d_out, int out_size, void* d_ws, size_t ws_size,
                              hipStream_t stream) {
    (void)in_sizes; (void)n_in; (void)out_size; (void)ws_size;
    const int Bn = 4, T = 2048, H = 1024, M = Bn * T;   // 8192

    const float* x  = (const float*)d_in[0];
    const float* Wk = (const float*)d_in[1];  const float* bk = (const float*)d_in[2];
    const float* Wq = (const float*)d_in[3];  const float* bq = (const float*)d_in[4];
    const float* Wv = (const float*)d_in[5];  const float* bv = (const float*)d_in[6];
    const float* W1 = (const float*)d_in[7];  const float* b1 = (const float*)d_in[8];
    const float* W2 = (const float*)d_in[9];  const float* b2 = (const float*)d_in[10];

    char* ws = (char*)d_ws;
    const size_t MB = 1024ull * 1024ull;
    unsigned short* xb  = (unsigned short*)(ws + 0);         // 16 MB
    unsigned short* wkT = (unsigned short*)(ws + 16 * MB);
    unsigned short* wqT = (unsigned short*)(ws + 18 * MB);
    unsigned short* wvT = (unsigned short*)(ws + 20 * MB);
    unsigned short* w1T = (unsigned short*)(ws + 22 * MB);
    unsigned short* w2T = (unsigned short*)(ws + 24 * MB);
    unsigned short* qb  = (unsigned short*)(ws + 26 * MB);   // 16 MB
    unsigned short* kb  = (unsigned short*)(ws + 42 * MB);   // 16 MB
    unsigned short* vt  = (unsigned short*)(ws + 58 * MB);   // 16 MB; end @ 74 MB

    unsigned short* attn_out = qb;   // qb_b dead before attn_out_b is written
    unsigned short* h1       = xb;   // xb dead after q/k/v projections

    float*          scores = (float*)d_out;                             // 16 MB
    unsigned short* attn   = (unsigned short*)((char*)d_out + 16 * MB); // 8 MB
    float*          outf   = (float*)d_out;

    // --- convert inputs to bf16 (+ transposed weights) ---
    cvt_f32_bf16<<<(M * H + 255) / 256, 256, 0, stream>>>(x, xb, M * H);
    transpose_cvt_1024<<<4096, 256, 0, stream>>>(Wk, wkT);
    transpose_cvt_1024<<<4096, 256, 0, stream>>>(Wq, wqT);
    transpose_cvt_1024<<<4096, 256, 0, stream>>>(Wv, wvT);
    transpose_cvt_1024<<<4096, 256, 0, stream>>>(W1, w1T);
    transpose_cvt_1024<<<4096, 256, 0, stream>>>(W2, w2T);

    // --- q/k/v projections ---
    dim3 gProj(H / 128, M / 256);   // (8, 32)
    gemm_bf16_wmma<EPI_BF16><<<gProj, 256, 0, stream>>>(
        xb, wqT, M, H, H, 1.0f, bq, nullptr, qb, nullptr);
    gemm_bf16_wmma<EPI_BF16><<<gProj, 256, 0, stream>>>(
        xb, wkT, M, H, H, 1.0f, bk, nullptr, kb, nullptr);
    gemm_bf16_wmma<EPI_BF16_VT><<<gProj, 256, 0, stream>>>(
        xb, wvT, M, H, H, 1.0f, bv, nullptr, vt, nullptr);

    // --- attention, one batch at a time (scores/attn scratch in d_out) ---
    const float inv_sqrt_h = 0.03125f;   // 1/sqrt(1024)
    dim3 gScore(T / 128, T / 256);       // (16, 8)
    dim3 gAV(H / 128, T / 256);          // (8, 8)
    for (int b = 0; b < Bn; ++b) {
        const size_t off = (size_t)b * T * H;
        gemm_bf16_wmma<EPI_F32_SCALE><<<gScore, 256, 0, stream>>>(
            qb + off, kb + off, T, T, H, inv_sqrt_h,
            nullptr, nullptr, nullptr, scores);
        softmax2048<<<T, 256, 0, stream>>>(scores, attn);
        gemm_bf16_wmma<EPI_BF16_NOBIAS><<<gAV, 256, 0, stream>>>(
            attn, vt + off, T, H, T, 1.0f, nullptr, nullptr,
            attn_out + off, nullptr);
    }

    // --- FFN: h1 = gelu(attn_out @ W1 + b1);  out = x + h1 @ W2 + b2 ---
    gemm_bf16_wmma<EPI_BF16_GELU><<<gProj, 256, 0, stream>>>(
        attn_out, w1T, M, H, H, 1.0f, b1, nullptr, h1, nullptr);
    gemm_bf16_wmma<EPI_F32_RESID><<<gProj, 256, 0, stream>>>(
        h1, w2T, M, H, H, 1.0f, b2, x, nullptr, outf);
}
